// IMVTensorLSTM_44710609551939
// MI455X (gfx1250) — matrix operand order repaired and checked by reading, not compiled, not run on hardware
//
#include <hip/hip_runtime.h>

typedef __attribute__((ext_vector_type(16))) _Float16 v16h;
typedef __attribute__((ext_vector_type(8)))  _Float16 v8h;
typedef __attribute__((ext_vector_type(8)))  float    v8f;

#define B_  128
#define T_  512
#define D_  32
#define N_  64
#define CH_ 64        // timesteps of x staged per LDS chunk

// ---- branch-free fast transcendentals (CDNA5 has native V_TANH_F32) ----
__device__ __forceinline__ float fast_rcp(float v) {
#if __has_builtin(__builtin_amdgcn_rcpf)
    return __builtin_amdgcn_rcpf(v);
#else
    return 1.0f / v;
#endif
}

__device__ __forceinline__ float fast_tanh(float v) {
#if __has_builtin(__builtin_amdgcn_tanhf)
    return __builtin_amdgcn_tanhf(v);
#else
    float e = __expf(2.0f * v);
    return 1.0f - 2.0f * fast_rcp(e + 1.0f);
#endif
}

__device__ __forceinline__ float sigf(float v) {
    return fast_rcp(1.0f + __expf(-v));
}

// One block per d-slice, 256 threads = 8 wave32. Wave w owns batch rows
// [16w,16w+16): its A-fragments, C-tiles and h-row stores are wave-local, so
// the recurrence runs with NO per-step barriers. Attention state (S1,S2) and
// cell state c are register-resident in C-fragment layout; the attention
// score h.F_alpha is computed by an extra WMMA against a one-column B
// fragment. Barriers only at 64-step x-chunk refreshes.
__global__ __launch_bounds__(256)
void imv_lstm_kernel(const float* __restrict__ x,
                     const float* __restrict__ Uj, const float* __restrict__ Ui,
                     const float* __restrict__ Uf, const float* __restrict__ Uo,
                     const float* __restrict__ Wj, const float* __restrict__ Wi,
                     const float* __restrict__ Wf, const float* __restrict__ Wo,
                     const float* __restrict__ bj_, const float* __restrict__ bi_,
                     const float* __restrict__ bf_, const float* __restrict__ bo_,
                     const float* __restrict__ Fa,  const float* __restrict__ Fab_,
                     const float* __restrict__ Fbw, const float* __restrict__ Fbb,
                     const float* __restrict__ Phiw, const float* __restrict__ Phib,
                     float* __restrict__ mu_ws, float* __restrict__ bw_ws)
{
    __shared__ __align__(32) _Float16 sWt[4 * N_ * N_];   // 32 KB Wt[g][n][k] f16
    __shared__ __align__(16) _Float16 sH[B_ * N_];        // 16 KB h[b][n] f16
    __shared__ float sXc[CH_ * B_];                       // 32 KB x chunk [tl][b]
    __shared__ float sFaTmp[N_];

    const int d    = blockIdx.x;
    const int tid  = threadIdx.x;
    const int w    = tid >> 5;
    const int l    = tid & 31;
    const int half = l >> 4;
    const int lm   = l & 15;

    // ---- setup: weights f32 -> f16, n-major (B-fragment friendly) ----
    #pragma unroll
    for (int g = 0; g < 4; ++g) {
        const float* Wg = (g == 0) ? Wj : (g == 1) ? Wi : (g == 2) ? Wf : Wo;
        for (int idx = tid; idx < N_ * N_; idx += 256) {
            int i = idx >> 6, k = idx & 63;
            sWt[g * 4096 + k * 64 + i] = (_Float16)Wg[d * 4096 + idx];
        }
    }
    if (tid < N_) sFaTmp[tid] = Fa[d * N_ + tid];
    for (int idx = tid; idx < B_ * N_; idx += 256) sH[idx] = (_Float16)0.0f;
    __syncthreads();

    // per-lane U/bias constants: uu[tn][g], pp[tn][g] for n = 16*tn + lm
    float uu[4][4], pp[4][4];
    #pragma unroll
    for (int g = 0; g < 4; ++g) {
        const float* Ug = (g == 0) ? Uj : (g == 1) ? Ui : (g == 2) ? Uf : Uo;
        const float* Bg = (g == 0) ? bj_ : (g == 1) ? bi_ : (g == 2) ? bf_ : bo_;
        #pragma unroll
        for (int tn = 0; tn < 4; ++tn) {
            uu[tn][g] = Ug[d * N_ + tn * 16 + lm];
            pp[tn][g] = Bg[d * N_ + tn * 16 + lm];
        }
    }
    const float fab = Fab_[d];

    // F_alpha as a B fragment with only column N=0 populated
    v16h bFa0, bFa1;
    #pragma unroll
    for (int e = 0; e < 16; ++e) {
        int k = 16 * half + e;
        bFa0[e] = (_Float16)((lm == 0) ? sFaTmp[k]      : 0.0f);
        bFa1[e] = (_Float16)((lm == 0) ? sFaTmp[32 + k] : 0.0f);
    }

    // register-resident state in C-fragment layout
    float cc[4][8], s2[4][8], hv[4][8], s1[8];
    #pragma unroll
    for (int tn = 0; tn < 4; ++tn)
        #pragma unroll
        for (int r = 0; r < 8; ++r) { cc[tn][r] = 0.0f; s2[tn][r] = 0.0f; hv[tn][r] = 0.0f; }
    #pragma unroll
    for (int r = 0; r < 8; ++r) s1[r] = 0.0f;

    const int arow = (16 * w + lm) * N_;
    union AU { v16h v; v8h h[2]; };

    for (int t = 0; t < T_; ++t) {
        if ((t & (CH_ - 1)) == 0) {
            __syncthreads();                       // chunk fully consumed
            for (int idx = tid; idx < CH_ * B_; idx += 256) {
                int tl = idx >> 7, b = idx & 127;
                sXc[tl * B_ + b] = x[(b * T_ + t + tl) * D_ + d];
            }
            __syncthreads();                       // chunk visible
        }
        const int tl = t & (CH_ - 1);

        // A fragments = h(t-1), own rows only (wave-local, no barrier)
        AU a0u, a1u;
        a0u.h[0] = *(const v8h*)&sH[arow +      16 * half];
        a0u.h[1] = *(const v8h*)&sH[arow + 8  + 16 * half];
        a1u.h[0] = *(const v8h*)&sH[arow + 32 + 16 * half];
        a1u.h[1] = *(const v8h*)&sH[arow + 40 + 16 * half];

        // ---- attention update for h(t-1) via score WMMA (skip t==0) ----
        if (t > 0) {
            v8f sc = {};
            sc = __builtin_amdgcn_wmma_f32_16x16x32_f16(false, a0u.v, false, bFa0,
                                                        (short)0, sc, false, false);
            sc = __builtin_amdgcn_wmma_f32_16x16x32_f16(false, a1u.v, false, bFa1,
                                                        (short)0, sc, false, false);
            #pragma unroll
            for (int r = 0; r < 8; ++r) {
                float s  = __shfl(sc[r], l & 16, 32) + fab;   // score at N=0 lanes
                float wt = __expf(fast_tanh(s));
                s1[r] += wt;
                #pragma unroll
                for (int tn = 0; tn < 4; ++tn) s2[tn][r] += wt * hv[tn][r];
            }
        }

        float xr[8];
        #pragma unroll
        for (int r = 0; r < 8; ++r) xr[r] = sXc[tl * B_ + 16 * w + r + 8 * half];

        // ---- 4 gate GEMMs: 16x16x32 WMMA over K=64 ----
        #pragma unroll
        for (int tn = 0; tn < 4; ++tn) {
            const int n = tn * 16 + lm;
            v8f acc[4];
            #pragma unroll
            for (int g = 0; g < 4; ++g) {
                const _Float16* wp = &sWt[g * 4096 + n * 64 + 16 * half];
                v16h b0 = *(const v16h*)(wp);
                v16h b1 = *(const v16h*)(wp + 32);
                v8f z = {};
                z = __builtin_amdgcn_wmma_f32_16x16x32_f16(false, a0u.v, false, b0,
                                                           (short)0, z, false, false);
                z = __builtin_amdgcn_wmma_f32_16x16x32_f16(false, a1u.v, false, b1,
                                                           (short)0, z, false, false);
                acc[g] = z;
            }
            #pragma unroll
            for (int r = 0; r < 8; ++r) {
                float jg = fast_tanh(acc[0][r] + xr[r] * uu[tn][0] + pp[tn][0]);
                float ig = sigf     (acc[1][r] + xr[r] * uu[tn][1] + pp[tn][1]);
                float fg = sigf     (acc[2][r] + xr[r] * uu[tn][2] + pp[tn][2]);
                float og = sigf     (acc[3][r] + xr[r] * uu[tn][3] + pp[tn][3]);
                float cn = cc[tn][r] * fg + ig * jg;
                cc[tn][r] = cn;
                float h  = og * fast_tanh(cn);
                hv[tn][r] = h;
                sH[(16 * w + r + 8 * half) * N_ + n] = (_Float16)h;   // own rows
            }
        }
    }

    // ---- tail attention update for h(T-1) ----
    {
        AU a0u, a1u;
        a0u.h[0] = *(const v8h*)&sH[arow +      16 * half];
        a0u.h[1] = *(const v8h*)&sH[arow + 8  + 16 * half];
        a1u.h[0] = *(const v8h*)&sH[arow + 32 + 16 * half];
        a1u.h[1] = *(const v8h*)&sH[arow + 40 + 16 * half];
        v8f sc = {};
        sc = __builtin_amdgcn_wmma_f32_16x16x32_f16(false, a0u.v, false, bFa0,
                                                    (short)0, sc, false, false);
        sc = __builtin_amdgcn_wmma_f32_16x16x32_f16(false, a1u.v, false, bFa1,
                                                    (short)0, sc, false, false);
        #pragma unroll
        for (int r = 0; r < 8; ++r) {
            float s  = __shfl(sc[r], l & 16, 32) + fab;
            float wt = __expf(fast_tanh(s));
            s1[r] += wt;
            #pragma unroll
            for (int tn = 0; tn < 4; ++tn) s2[tn][r] += wt * hv[tn][r];
        }
    }

    // ---- final: mu = hg.Phi_w + Phi_b ; beta = exp(tanh(hg.Fbw + Fbb)) ----
    float pw0[4], pw1[4], fw0[4], fw1[4];
    #pragma unroll
    for (int tn = 0; tn < 4; ++tn) {
        int n = tn * 16 + lm;
        pw0[tn] = Phiw[n];  pw1[tn] = Phiw[64 + n];
        fw0[tn] = Fbw[n];   fw1[tn] = Fbw[64 + n];
    }
    const float phib = Phib[0], fbb = Fbb[0];
    #pragma unroll
    for (int r = 0; r < 8; ++r) {
        float inv = fast_rcp(s1[r]);
        float pmu = 0.0f, pbl = 0.0f;
        #pragma unroll
        for (int tn = 0; tn < 4; ++tn) {
            float gn = s2[tn][r] * inv;            // g_n
            pmu += gn * pw0[tn] + hv[tn][r] * pw1[tn];
            pbl += gn * fw0[tn] + hv[tn][r] * fw1[tn];
        }
        #pragma unroll
        for (int m = 1; m < 16; m <<= 1) {         // reduce over 16-lane group
            pmu += __shfl_xor(pmu, m, 32);
            pbl += __shfl_xor(pbl, m, 32);
        }
        if (lm == 0) {
            int b = 16 * w + r + 8 * half;
            mu_ws[b * D_ + d] = pmu + phib;
            bw_ws[b * D_ + d] = __expf(fast_tanh(pbl + fbb));
        }
    }
}

// Reduce across D: out[b] = sum_d beta*mu / sum_d beta
__global__ __launch_bounds__(128)
void imv_finalize_kernel(const float* __restrict__ mu_ws,
                         const float* __restrict__ bw_ws,
                         float* __restrict__ out)
{
    const int b = threadIdx.x;
    float num = 0.0f, den = 0.0f;
    #pragma unroll
    for (int d = 0; d < D_; ++d) {
        float wv = bw_ws[b * D_ + d];
        num += wv * mu_ws[b * D_ + d];
        den += wv;
    }
    out[b] = num * fast_rcp(den);
}

extern "C" void kernel_launch(void* const* d_in, const int* in_sizes, int n_in,
                              void* d_out, int out_size, void* d_ws, size_t ws_size,
                              hipStream_t stream) {
    (void)in_sizes; (void)n_in; (void)out_size; (void)ws_size;
    const float* x    = (const float*)d_in[0];
    const float* Uj   = (const float*)d_in[1];
    const float* Ui   = (const float*)d_in[2];
    const float* Uf   = (const float*)d_in[3];
    const float* Uo   = (const float*)d_in[4];
    const float* Wj   = (const float*)d_in[5];
    const float* Wi   = (const float*)d_in[6];
    const float* Wf   = (const float*)d_in[7];
    const float* Wo   = (const float*)d_in[8];
    const float* bj   = (const float*)d_in[9];
    const float* bi   = (const float*)d_in[10];
    const float* bf   = (const float*)d_in[11];
    const float* bo   = (const float*)d_in[12];
    const float* Fa   = (const float*)d_in[13];
    const float* Fab  = (const float*)d_in[14];
    const float* Fbw  = (const float*)d_in[15];
    const float* Fbb  = (const float*)d_in[16];
    const float* Phiw = (const float*)d_in[17];
    const float* Phib = (const float*)d_in[18];

    float* mu_ws = (float*)d_ws;
    float* bw_ws = mu_ws + B_ * D_;

    imv_lstm_kernel<<<D_, 256, 0, stream>>>(x, Uj, Ui, Uf, Uo, Wj, Wi, Wf, Wo,
                                            bj, bi, bf, bo, Fa, Fab, Fbw, Fbb,
                                            Phiw, Phib, mu_ws, bw_ws);
    imv_finalize_kernel<<<1, B_, 0, stream>>>(mu_ws, bw_ws, (float*)d_out);
}
